// EncodeBlock_14791867367613
// MI455X (gfx1250) — compile-verified, matrix operand order checked by reading, not run on hardware
//
#include <hip/hip_runtime.h>

// ---------------- problem dims ----------------
static constexpr int BB = 8;
static constexpr int SS = 1024;
static constexpr int EE = 1024;
static constexpr int HH = 8;
static constexpr int DD = 128;   // EE / HH

typedef __attribute__((ext_vector_type(16))) __bf16 v16bf;
typedef __attribute__((ext_vector_type(8)))  float  v8f;

#define DEV static __device__ __forceinline__

union BF16Frag { v16bf v; unsigned int u[8]; unsigned short s[16]; };
union BFBits   { __bf16 b; unsigned short s; };

DEV unsigned short f2bf(float f) {      // native cvt path (RNE)
  BFBits x; x.b = (__bf16)f;
  return x.s;
}
DEV float bf2f(unsigned short h) { return __uint_as_float(((unsigned int)h) << 16); }

DEV v8f zero8() {
  v8f z;
#pragma unroll
  for (int i = 0; i < 8; ++i) z[i] = 0.0f;
  return z;
}

DEV v8f wmma_bf16(v16bf a, v16bf b, v8f c) {
  // D = A(16x32 bf16) * B(32x16 bf16) + C(16x16 f32)
  return __builtin_amdgcn_wmma_f32_16x16x32_bf16(false, a, false, b, (short)0, c,
                                                 false, false);
}

// CDNA5 async copy: global -> LDS, 16B per lane, tracked by ASYNCcnt.
DEV void async_load_b128(unsigned lds_addr, const void* gaddr) {
  asm volatile("global_load_async_to_lds_b128 %0, %1, off"
               :: "v"(lds_addr), "v"(gaddr) : "memory");
}
DEV void wait_asynccnt0() { asm volatile("s_wait_asynccnt 0" ::: "memory"); }

// A-fragment (16x32, row = lane%16) or B-fragment (32x16, col = lane%16) loader.
// `base` points at the 32 k-contiguous bf16 elements of this lane's row/col.
// ISA layout: VGPR r<4 holds K = 2r + kb, r>=4 holds K = 16 + 2(r-4) + kb,
// kb = 8 for lanes 16..31 (pair offset 4).
DEV v16bf load_frag_k32(const unsigned short* base, int lane) {
  const unsigned int* p = (const unsigned int*)base;
  const int off = (lane >> 4) << 2;
  BF16Frag f;
#pragma unroll
  for (int r = 0; r < 4; ++r) {
    f.u[r]     = p[r + off];
    f.u[4 + r] = p[8 + r + off];
  }
  return f.v;
}

// B-fragment from an fp32 matrix with K-stride `ld` (element [k][n] = base[k*ld]).
DEV v16bf load_bfrag_strided_f32(const float* base, int ld, int lane) {
  const int kb = (lane >> 4) * 8;
  BF16Frag f;
#pragma unroll
  for (int r = 0; r < 4; ++r) {
    int k0 = 2 * r + kb;
    f.s[2 * r]     = f2bf(base[(size_t)k0 * ld]);
    f.s[2 * r + 1] = f2bf(base[(size_t)(k0 + 1) * ld]);
    int k1 = 16 + 2 * r + kb;
    f.s[8 + 2 * r]     = f2bf(base[(size_t)k1 * ld]);
    f.s[8 + 2 * r + 1] = f2bf(base[(size_t)(k1 + 1) * ld]);
  }
  return f.v;
}

DEV v16bf scale_frag(v16bf a, float sc) {
  BF16Frag f; f.v = a;
#pragma unroll
  for (int i = 0; i < 16; ++i) f.s[i] = f2bf(bf2f(f.s[i]) * sc);
  return f.v;
}

DEV float head_logk(int h) {
  float t  = (float)h * (1.0f / (float)(HH - 1));
  float lo = logf(1.0f / 32.0f), hi = logf(1.0f / 512.0f);
  float kappa = 1.0f - expf(lo + (hi - lo) * t);
  return logf(kappa);
}

DEV float wave_sum(float v) {
#pragma unroll
  for (int m = 16; m >= 1; m >>= 1) v += __shfl_xor(v, m, 32);
  return v;
}

// ---------------- elementwise / scan kernels ----------------

__global__ void f32_to_bf16_kernel(const float* __restrict__ in,
                                   unsigned short* __restrict__ out, size_t n) {
  size_t i = (size_t)blockIdx.x * blockDim.x + threadIdx.x;
  size_t st = (size_t)gridDim.x * blockDim.x;
  for (; i < n; i += st) out[i] = f2bf(in[i]);
}

__global__ void seg_scan_kernel(const unsigned char* __restrict__ dones,
                                int* __restrict__ seg) {
  if (threadIdx.x != 0) return;
  int b = blockIdx.x;
  int acc = 0;
  for (int s = 0; s < SS; ++s) {
    acc += dones[(size_t)b * SS + s] ? 1 : 0;
    seg[(size_t)b * SS + s] = acc;
  }
}

__global__ void silumul_kernel(const float* __restrict__ g,
                               const float* __restrict__ l,
                               unsigned short* __restrict__ out, size_t n) {
  size_t i = (size_t)blockIdx.x * blockDim.x + threadIdx.x;
  size_t st = (size_t)gridDim.x * blockDim.x;
  for (; i < n; i += st) {
    float gv = g[i];
    out[i] = f2bf((gv / (1.0f + expf(-gv))) * l[i]);
  }
}

// Per-(b,s,h) groupnorm over D=128 + silu(g) gating -> bf16 A for the w_o GEMM.
__global__ __launch_bounds__(256) void gnorm_gate_kernel(
    const float* __restrict__ Ret, const unsigned short* __restrict__ G,
    const float* __restrict__ gn_scale, const float* __restrict__ gn_bias,
    unsigned short* __restrict__ RS) {
  int row = blockIdx.x;                    // b*S + s
  int lane = threadIdx.x & 31, h = threadIdx.x >> 5;   // wave per head
  const float* rp = Ret + (size_t)row * EE + h * DD;
  float vals[4], sum = 0.f, sq = 0.f;
#pragma unroll
  for (int i = 0; i < 4; ++i) {
    float v = rp[lane * 4 + i];
    vals[i] = v; sum += v; sq += v * v;
  }
  sum = wave_sum(sum); sq = wave_sum(sq);
  float mean = sum * (1.0f / DD);
  float var  = sq * (1.0f / DD) - mean * mean;
  float rstd = rsqrtf(var + 1e-5f);
#pragma unroll
  for (int i = 0; i < 4; ++i) {
    int e = h * DD + lane * 4 + i;
    float nv = (vals[i] - mean) * rstd * gn_scale[e] + gn_bias[e];
    float gv = bf2f(G[(size_t)row * EE + e]);
    float sil = gv / (1.0f + expf(-gv));
    RS[(size_t)row * EE + e] = f2bf(nv * sil);
  }
}

// out = rmsnorm(X + Y, scale); optional fp32 + bf16 outputs.
__global__ __launch_bounds__(256) void resid_rmsnorm_kernel(
    const float* __restrict__ X, const float* __restrict__ Y,
    const float* __restrict__ scale, float* __restrict__ o32,
    unsigned short* __restrict__ o16) {
  __shared__ float red[8];
  int row = blockIdx.x, tid = threadIdx.x, lane = tid & 31, wid = tid >> 5;
  float v[4], sq = 0.f;
#pragma unroll
  for (int i = 0; i < 4; ++i) {
    int e = tid * 4 + i;
    float t = X[(size_t)row * EE + e] + Y[(size_t)row * EE + e];
    v[i] = t; sq += t * t;
  }
  sq = wave_sum(sq);
  if (lane == 0) red[wid] = sq;
  __syncthreads();
  float tot = 0.f;
#pragma unroll
  for (int w = 0; w < 8; ++w) tot += red[w];
  float r = rsqrtf(tot * (1.0f / EE) + 1e-6f);
#pragma unroll
  for (int i = 0; i < 4; ++i) {
    int e = tid * 4 + i;
    float o = v[i] * r * scale[e];
    if (o32) o32[(size_t)row * EE + e] = o;
    if (o16) o16[(size_t)row * EE + e] = f2bf(o);
  }
}

// ---------------- tiled bf16 WMMA GEMM (async + double-buffered) ----------------
// C[M,N] = A[M,K] * W[K,N]; A,W bf16; 128x128 block tile, 8 waves, K-step 32.
// A tiles: global_load_async_to_lds_b128 (ASYNCcnt), ping-pong LDS buffers.
// B tiles: register-staged (needs element transpose), issued before compute.
template <int STORE_BF16>
__global__ __launch_bounds__(256) void gemm_bf16(
    const unsigned short* __restrict__ A, const unsigned short* __restrict__ W,
    float* __restrict__ Cf, unsigned short* __restrict__ Cb,
    int M, int N, int K) {
  __shared__ unsigned short As[2][128 * 32];   // [m][k] row-major
  __shared__ unsigned short Bs[2][128 * 32];   // [n][k] (W transposed, k-contig)
  int tid = threadIdx.x, lane = tid & 31, wid = tid >> 5;
  int m0 = blockIdx.y * 128, n0 = blockIdx.x * 128;
  int baseM = wid * 16;
  v8f acc[8];
#pragma unroll
  for (int i = 0; i < 8; ++i) acc[i] = zero8();

  unsigned short bReg[16];
  auto issue_async_A = [&](int k0, int buf) {
    unsigned base = (unsigned)(unsigned long long)(&As[buf][0]);
#pragma unroll
    for (int i = 0; i < 2; ++i) {
      int idx = tid + 256 * i;          // 512 x 16B chunks
      int m = idx >> 2, c = idx & 3;    // 4 chunks (64B) per row
      async_load_b128(base + idx * 16,
                      A + (size_t)(m0 + m) * K + k0 + c * 8);
    }
  };
  auto load_B_regs = [&](int k0) {
#pragma unroll
    for (int i = 0; i < 16; ++i) {
      int idx = tid + 256 * i;
      int n = idx & 127, k = idx >> 7;
      bReg[i] = W[(size_t)(k0 + k) * N + n0 + n];
    }
  };

  issue_async_A(0, 0);
  load_B_regs(0);

  for (int k0 = 0, it = 0; k0 < K; k0 += 32, ++it) {
    int p = it & 1;
    // commit this tile's B registers to LDS
#pragma unroll
    for (int i = 0; i < 16; ++i) {
      int idx = tid + 256 * i;
      Bs[p][(idx & 127) * 32 + (idx >> 7)] = bReg[i];
    }
    wait_asynccnt0();                     // our async A writes for tile `it`
    __syncthreads();                      // all waves' LDS writes visible
    if (k0 + 32 < K) {                    // start next tile before compute
      issue_async_A(k0 + 32, p ^ 1);
      load_B_regs(k0 + 32);
    }

    v16bf af = load_frag_k32(&As[p][(baseM + (lane & 15)) * 32], lane);
#pragma unroll
    for (int g = 0; g < 2; ++g) {
      v16bf b4[4];
#pragma unroll
      for (int t = 0; t < 4; ++t)
        b4[t] = load_frag_k32(&Bs[p][((g * 4 + t) * 16 + (lane & 15)) * 32], lane);
#pragma unroll
      for (int t = 0; t < 4; ++t)
        acc[g * 4 + t] = wmma_bf16(af, b4[t], acc[g * 4 + t]);
    }
  }

  int cl = lane & 15, rh = (lane >> 4) * 8;
#pragma unroll
  for (int nt = 0; nt < 8; ++nt)
#pragma unroll
    for (int r = 0; r < 8; ++r) {
      size_t o = (size_t)(m0 + baseM + r + rh) * N + (n0 + nt * 16 + cl);
      if (STORE_BF16) Cb[o] = f2bf(acc[nt][r]);
      else            Cf[o] = acc[nt][r];
    }
}

// ---------------- retention (scores*decay)*v + cross(q,xi,hstate) ----------------
__global__ __launch_bounds__(256) void retention_kernel(
    const unsigned short* __restrict__ Q, const unsigned short* __restrict__ Kq,
    const unsigned short* __restrict__ V, const float* __restrict__ Hst,
    const int* __restrict__ ts, const int* __restrict__ seg,
    float* __restrict__ Ret) {
  __shared__ unsigned short Ksl[32 * 128];        // [j][d]   (B for scores)
  __shared__ unsigned short Vsl[128 * 32];        // [e][j]   (B for inner)
  __shared__ unsigned short Ssl[8 * 16 * 32];     // per-wave score scratch [m][j]

  int tid = threadIdx.x, lane = tid & 31, wid = tid >> 5;
  int b = blockIdx.y / HH, h = blockIdx.y % HH;
  int iw = blockIdx.x * 128 + wid * 16;           // this wave's 16 query rows
  int cl = lane & 15, hh = lane >> 4;

  const float logk = head_logk(h);
  const float kscale = 0.0883883476483f;          // D^-0.5

  // C-layout row metadata
  float tsi[8]; int segi[8];
#pragma unroll
  for (int r = 0; r < 8; ++r) {
    int row = iw + r + 8 * hh;
    tsi[r]  = (float)ts[(size_t)b * SS + row];
    segi[r] = seg[(size_t)b * SS + row];
  }
  // A-layout row metadata
  int arow = iw + cl;
  float tsa = (float)ts[(size_t)b * SS + arow];
  int   sga = seg[(size_t)b * SS + arow];

  // q A-fragments for the 4 K-steps over D=128
  v16bf qa[4];
  const unsigned short* qrow = Q + ((size_t)b * SS + arow) * EE + h * DD;
#pragma unroll
  for (int c = 0; c < 4; ++c) qa[c] = load_frag_k32(qrow + c * 32, lane);

  v8f acc[8];
#pragma unroll
  for (int i = 0; i < 8; ++i) acc[i] = zero8();

  unsigned short* sr = &Ssl[wid * 512];

  for (int j0 = 0; j0 < SS; j0 += 32) {
    __syncthreads();
    const unsigned int* Kg = (const unsigned int*)Kq;
#pragma unroll
    for (int i = 0; i < 8; ++i) {         // K tile [j][d], dword copies
      int idx = tid + 256 * i;
      int j = idx >> 6, dp = idx & 63;
      ((unsigned int*)Ksl)[idx] =
          Kg[(((size_t)b * SS + j0 + j) * EE + h * DD) / 2 + dp];
    }
#pragma unroll
    for (int i = 0; i < 16; ++i) {        // V^T tile [e][j]
      int idx = tid + 256 * i;
      int e = idx & 127, j = idx >> 7;
      Vsl[e * 32 + j] = V[((size_t)b * SS + j0 + j) * EE + h * DD + e];
    }
    __syncthreads();

    float tsj0 = (float)ts[(size_t)b * SS + j0 + cl];
    float tsj1 = (float)ts[(size_t)b * SS + j0 + 16 + cl];
    int   sj0  = seg[(size_t)b * SS + j0 + cl];
    int   sj1  = seg[(size_t)b * SS + j0 + 16 + cl];

    v8f s0 = zero8(), s1 = zero8();
#pragma unroll
    for (int c = 0; c < 4; ++c) {         // scores = q . k^T over D=128
      v16bf b0 = load_frag_k32(&Ksl[cl * 128 + c * 32], lane);
      v16bf b1 = load_frag_k32(&Ksl[(16 + cl) * 128 + c * 32], lane);
      s0 = wmma_bf16(qa[c], b0, s0);
      s1 = wmma_bf16(qa[c], b1, s1);
    }
    // decay mask on fp32 scores, then stash as bf16 A-matrix in LDS
#pragma unroll
    for (int r = 0; r < 8; ++r) {
      int mrow = r + 8 * hh;
      float d0 = (segi[r] == sj0) ? expf(logk * fabsf(tsi[r] - tsj0)) * kscale : 0.f;
      float d1 = (segi[r] == sj1) ? expf(logk * fabsf(tsi[r] - tsj1)) * kscale : 0.f;
      sr[mrow * 32 + cl]      = f2bf(s0[r] * d0);
      sr[mrow * 32 + 16 + cl] = f2bf(s1[r] * d1);
    }
    __syncthreads();

    v16bf afr = load_frag_k32(&sr[cl * 32], lane);
#pragma unroll
    for (int g = 0; g < 2; ++g) {         // inner += scores . v  (batched B loads)
      v16bf b4[4];
#pragma unroll
      for (int t = 0; t < 4; ++t)
        b4[t] = load_frag_k32(&Vsl[((g * 4 + t) * 16 + cl) * 32], lane);
#pragma unroll
      for (int t = 0; t < 4; ++t)
        acc[g * 4 + t] = wmma_bf16(afr, b4[t], acc[g * 4 + t]);
    }
  }

  // cross = (q * xi) . hstate
  float xi = (sga == 0) ? expf(logk * (tsa + 1.0f)) : 0.0f;
  v16bf qx[4];
#pragma unroll
  for (int c = 0; c < 4; ++c) qx[c] = scale_frag(qa[c], xi);
  const float* Hb = Hst + (size_t)(b * HH + h) * DD * DD;
#pragma unroll
  for (int nt = 0; nt < 8; ++nt)
#pragma unroll
    for (int c = 0; c < 4; ++c) {
      v16bf hb = load_bfrag_strided_f32(Hb + (size_t)(c * 32) * DD + nt * 16 + cl,
                                        DD, lane);
      acc[nt] = wmma_bf16(qx[c], hb, acc[nt]);
    }

#pragma unroll
  for (int nt = 0; nt < 8; ++nt)
#pragma unroll
    for (int r = 0; r < 8; ++r) {
      int srow = iw + r + 8 * hh;
      Ret[((size_t)b * SS + srow) * EE + h * DD + nt * 16 + cl] = acc[nt][r];
    }
}

// ---------------- next_h = chunk_decay*hstate*carry + (k*delta)^T . v ----------------
__global__ __launch_bounds__(256) void nexth_kernel(
    const unsigned short* __restrict__ Kq, const unsigned short* __restrict__ V,
    const float* __restrict__ Hst, const int* __restrict__ ts,
    const int* __restrict__ seg, float* __restrict__ OutH) {
  __shared__ unsigned short KT[128 * 32];   // [d][s] (A, k-contiguous = s)
  __shared__ unsigned short VT[128 * 32];   // [e][s] (B, k-contiguous = s)
  int tid = threadIdx.x, lane = tid & 31, wid = tid >> 5;
  int b = blockIdx.x / HH, h = blockIdx.x % HH;
  int cl = lane & 15, hh = lane >> 4;

  const float logk = head_logk(h);
  const float kscale = 0.0883883476483f;
  float tsl = (float)ts[(size_t)b * SS + SS - 1];
  int   sgl = seg[(size_t)b * SS + SS - 1];

  v8f acc[8];
#pragma unroll
  for (int i = 0; i < 8; ++i) acc[i] = zero8();

  for (int s0 = 0; s0 < SS; s0 += 32) {
    __syncthreads();
#pragma unroll
    for (int i = 0; i < 16; ++i) {
      int idx = tid + 256 * i;
      int d = idx & 127, s = idx >> 7;
      int gs = s0 + s;
      float delta = (seg[(size_t)b * SS + gs] == sgl)
                        ? expf(logk * (tsl - (float)ts[(size_t)b * SS + gs]))
                        : 0.0f;
      float kv = bf2f(Kq[((size_t)b * SS + gs) * EE + h * DD + d]) * kscale * delta;
      KT[d * 32 + s] = f2bf(kv);
      VT[d * 32 + s] = V[((size_t)b * SS + gs) * EE + h * DD + d];
    }
    __syncthreads();
    v16bf af = load_frag_k32(&KT[(wid * 16 + cl) * 32], lane);
#pragma unroll
    for (int g = 0; g < 2; ++g) {
      v16bf b4[4];
#pragma unroll
      for (int t = 0; t < 4; ++t)
        b4[t] = load_frag_k32(&VT[((g * 4 + t) * 16 + cl) * 32], lane);
#pragma unroll
      for (int t = 0; t < 4; ++t)
        acc[g * 4 + t] = wmma_bf16(af, b4[t], acc[g * 4 + t]);
    }
  }

  float chunk = expf(logk * (tsl + 1.0f));
  float carry = (sgl == 0) ? 1.0f : 0.0f;
#pragma unroll
  for (int nt = 0; nt < 8; ++nt)
#pragma unroll
    for (int r = 0; r < 8; ++r) {
      int d = wid * 16 + r + 8 * hh, e = nt * 16 + cl;
      size_t o = (size_t)(b * HH + h) * DD * DD + (size_t)d * DD + e;
      OutH[o] = chunk * carry * Hst[o] + acc[nt][r];
    }
}

// ---------------- host-side orchestration ----------------
extern "C" void kernel_launch(void* const* d_in, const int* in_sizes, int n_in,
                              void* d_out, int out_size, void* d_ws, size_t ws_size,
                              hipStream_t stream) {
  (void)in_sizes; (void)n_in; (void)out_size; (void)ws_size;
  const float* x        = (const float*)d_in[0];
  const float* hstate   = (const float*)d_in[1];
  const float* w_q      = (const float*)d_in[2];
  const float* w_k      = (const float*)d_in[3];
  const float* w_v      = (const float*)d_in[4];
  const float* w_g      = (const float*)d_in[5];
  const float* w_o      = (const float*)d_in[6];
  const float* gn_scale = (const float*)d_in[7];
  const float* gn_bias  = (const float*)d_in[8];
  const float* ln1      = (const float*)d_in[9];
  const float* ln2      = (const float*)d_in[10];
  const float* w_fg     = (const float*)d_in[11];
  const float* w_fl     = (const float*)d_in[12];
  const float* w_fo     = (const float*)d_in[13];
  const unsigned char* dones = (const unsigned char*)d_in[14];
  const int* stepc      = (const int*)d_in[15];

  float* out_main = (float*)d_out;                          // [B,S,E] f32
  float* out_h    = out_main + (size_t)BB * SS * EE;        // [B,H,D,D] f32

  const size_t NBSE = (size_t)BB * SS * EE;   // 8388608
  const size_t NEE  = (size_t)EE * EE;        // 1048576

  char* wsp = (char*)d_ws;
  auto alloc = [&](size_t bytes) {
    char* p = wsp;
    wsp += (bytes + 255) & ~(size_t)255;
    return p;
  };
  unsigned short* x_bf = (unsigned short*)alloc(NBSE * 2);
  unsigned short* wb[8];
  for (int i = 0; i < 8; ++i) wb[i] = (unsigned short*)alloc(NEE * 2);
  unsigned short* q_bf = (unsigned short*)alloc(NBSE * 2);
  unsigned short* k_bf = (unsigned short*)alloc(NBSE * 2);
  unsigned short* v_bf = (unsigned short*)alloc(NBSE * 2);
  unsigned short* g_bf = (unsigned short*)alloc(NBSE * 2);
  float* ret_f  = (float*)alloc(NBSE * 4);
  unsigned short* rs_bf = (unsigned short*)alloc(NBSE * 2);
  float* attn_f = (float*)alloc(NBSE * 4);
  float* h_f    = (float*)alloc(NBSE * 4);
  int*   seg    = (int*)alloc((size_t)BB * SS * 4);
  // Safe aliases (producers run strictly after the aliased buffer is consumed;
  // all kernels serialize on `stream`):
  float* gate_f = attn_f;                    // attn consumed by ln1 kernel
  float* lin_f  = (float*)q_bf;              // spans q_bf+k_bf (32MB contiguous)
  unsigned short* act_bf = v_bf;             // v consumed by retention/nexth
  unsigned short* hb_bf  = g_bf;             // g consumed by groupnorm gate
  float* ffn_f  = ret_f;                     // ret consumed by groupnorm gate

  dim3 blk(256);
  dim3 ggrid(EE / 128, (BB * SS) / 128);     // (8, 64)
  dim3 rgrid(SS / 128, BB * HH);             // (8, 64)

  // 1) precision staging + segment scan
  f32_to_bf16_kernel<<<2048, blk, 0, stream>>>(x, x_bf, NBSE);
  const float* wsrc[8] = {w_q, w_k, w_v, w_g, w_o, w_fg, w_fl, w_fo};
  for (int i = 0; i < 8; ++i)
    f32_to_bf16_kernel<<<512, blk, 0, stream>>>(wsrc[i], wb[i], NEE);
  seg_scan_kernel<<<BB, 32, 0, stream>>>(dones, seg);

  // 2) q, k, v, g projections (bf16 out)
  gemm_bf16<1><<<ggrid, blk, 0, stream>>>(x_bf, wb[0], nullptr, q_bf, BB * SS, EE, EE);
  gemm_bf16<1><<<ggrid, blk, 0, stream>>>(x_bf, wb[1], nullptr, k_bf, BB * SS, EE, EE);
  gemm_bf16<1><<<ggrid, blk, 0, stream>>>(x_bf, wb[2], nullptr, v_bf, BB * SS, EE, EE);
  gemm_bf16<1><<<ggrid, blk, 0, stream>>>(x_bf, wb[3], nullptr, g_bf, BB * SS, EE, EE);

  // 3) retention core + recurrent state
  retention_kernel<<<rgrid, blk, 0, stream>>>(q_bf, k_bf, v_bf, hstate, stepc, seg, ret_f);
  nexth_kernel<<<BB * HH, blk, 0, stream>>>(k_bf, v_bf, hstate, stepc, seg, out_h);

  // 4) groupnorm + silu gate, then output projection
  gnorm_gate_kernel<<<BB * SS, blk, 0, stream>>>(ret_f, g_bf, gn_scale, gn_bias, rs_bf);
  gemm_bf16<0><<<ggrid, blk, 0, stream>>>(rs_bf, wb[4], attn_f, nullptr, BB * SS, EE, EE);

  // 5) h = rmsnorm(x + attn, ln1)
  resid_rmsnorm_kernel<<<BB * SS, blk, 0, stream>>>(x, attn_f, ln1, h_f, hb_bf);

  // 6) FFN
  gemm_bf16<0><<<ggrid, blk, 0, stream>>>(hb_bf, wb[5], gate_f, nullptr, BB * SS, EE, EE);
  gemm_bf16<0><<<ggrid, blk, 0, stream>>>(hb_bf, wb[6], lin_f, nullptr, BB * SS, EE, EE);
  silumul_kernel<<<2048, blk, 0, stream>>>(gate_f, lin_f, act_bf, NBSE);
  gemm_bf16<0><<<ggrid, blk, 0, stream>>>(act_bf, wb[7], ffn_f, nullptr, BB * SS, EE, EE);

  // 7) final rmsnorm(h + ffn, ln2) -> d_out
  resid_rmsnorm_kernel<<<BB * SS, blk, 0, stream>>>(h_f, ffn_f, ln2, out_main, nullptr);
}
